// AttentionLayer_27496380629625
// MI455X (gfx1250) — matrix operand order repaired and checked by reading, not compiled, not run on hardware
//
#include <hip/hip_runtime.h>
#include <hip/hip_fp16.h>

typedef __attribute__((ext_vector_type(16))) _Float16 v16h;
typedef __attribute__((ext_vector_type(8)))  float    v8f;
typedef unsigned int v4u_t __attribute__((ext_vector_type(4)));
typedef int          v4i_t __attribute__((ext_vector_type(4)));
typedef int          v8i_t __attribute__((ext_vector_type(8)));

#define B_    4
#define S_    2048
#define DIN   1024
#define DOUT  1024
#define H_    16
#define DH_   64
#define N3    (3 * DOUT)

union AFrag { v16h v; _Float16 e[16]; uint4 q[2]; };
union Acc   { v8f  v; float    e[8];  };

// ---- vectorized fragment builders (CDNA5 16-bit WMMA layouts, CDNA5 ISA §7.12.2) ----

// A 16x32 f16: lane row M = lane&15; elements [kh..kh+7],[16+kh..16+kh+7], kh=(lane>=16)?8:0.
__device__ inline v16h a_frag16(const _Float16* base, int stride, int lane) {
    AFrag f;
    const _Float16* rp = base + (lane & 15) * stride + (lane >> 4) * 8;
    f.q[0] = *(const uint4*)(rp);
    f.q[1] = *(const uint4*)(rp + 16);
    return f.v;
}

// B 32x16 f16: lane col N = lane&15; elements K [kh..kh+7],[16+kh..16+kh+7], kh=(lane>=16)?8:0. Source [n][k].
__device__ inline v16h b_frag16(const _Float16* base, int stride, int lane) {
    AFrag f;
    const _Float16* bp = base + (lane & 15) * stride + (lane >> 4) * 8;
    f.q[0] = *(const uint4*)(bp);
    f.q[1] = *(const uint4*)(bp + 16);
    return f.v;
}

__device__ inline v8f wmma_f16(v16h a, v16h b, v8f c) {
    return __builtin_amdgcn_wmma_f32_16x16x32_f16(false, a, false, b, (short)0, c,
                                                  false, false);
}

// =============================== Kernel 1: QKV projection ===============================
__global__ __launch_bounds__(256) void qkv_gemm(const float* __restrict__ payload,
                                                const float* __restrict__ wqkv,
                                                _Float16* __restrict__ qbuf,
                                                _Float16* __restrict__ kbuf,
                                                _Float16* __restrict__ vbuf) {
    constexpr int LDA = 48;  // 96B rows
    constexpr int LDB = 40;  // 80B per-n stride in transposed B tile
    __shared__ __align__(16) _Float16 sA[64 * LDA];
    __shared__ __align__(16) _Float16 sBt[64 * LDB];  // [n][k]

    const int t = threadIdx.x, lane = t & 31, wid = t >> 5;
    const int nbase = blockIdx.x * 64;
    const int mbase = blockIdx.y * 64;
    const int mt = wid & 3, nw = wid >> 2;

    Acc acc[2];
#pragma unroll
    for (int j = 0; j < 2; ++j)
#pragma unroll
        for (int i = 0; i < 8; ++i) acc[j].e[i] = 0.f;

    for (int k0 = 0; k0 < DIN; k0 += 32) {
        __syncthreads();
        // stage A 64x32 fp32->f16
#pragma unroll
        for (int it = 0; it < 2; ++it) {
            const int lin = t + it * 256;
            const int row = lin >> 3, c4 = (lin & 7) * 4;
            const float4 va = *(const float4*)(payload + (size_t)(mbase + row) * DIN + k0 + c4);
            sA[row * LDA + c4 + 0] = (_Float16)va.x;
            sA[row * LDA + c4 + 1] = (_Float16)va.y;
            sA[row * LDA + c4 + 2] = (_Float16)va.z;
            sA[row * LDA + c4 + 3] = (_Float16)va.w;
        }
        // stage B 32k x 64n transposed (coalesced reads, 8B LDS stores)
#pragma unroll
        for (int it = 0; it < 2; ++it) {
            const int nn = (t & 31) + it * 32;
            const int kg = (t >> 5) * 4;
#pragma unroll
            for (int j = 0; j < 4; ++j)
                sBt[nn * LDB + kg + j] =
                    (_Float16)wqkv[(size_t)(k0 + kg + j) * N3 + nbase + nn];
        }
        __syncthreads();
        if (k0 + 32 < DIN)
            __builtin_prefetch(payload + (size_t)(mbase + (t >> 3)) * DIN + k0 + 32, 0, 1);

        const v16h a  = a_frag16(sA + mt * 16 * LDA, LDA, lane);
        const v16h b0 = b_frag16(sBt + (nw * 32 + 0) * LDB, LDB, lane);
        const v16h b1 = b_frag16(sBt + (nw * 32 + 16) * LDB, LDB, lane);
        acc[0].v = wmma_f16(a, b0, acc[0].v);
        acc[1].v = wmma_f16(a, b1, acc[1].v);
    }

    const int col = lane & 15, rowoff = (lane >> 4) * 8;
    const int bb = mbase >> 11;
    const int srow = (mbase & (S_ - 1)) + mt * 16 + rowoff;
#pragma unroll
    for (int j = 0; j < 2; ++j) {
        const int gn = nbase + nw * 32 + j * 16 + col;
        const int sec = gn >> 10, w = gn & 1023, head = w >> 6, d = w & 63;
        if (sec == 2) {  // V transposed [b,h,d,s]
            _Float16* p = vbuf + (((size_t)bb * H_ + head) * DH_ + d) * S_ + srow;
#pragma unroll
            for (int r = 0; r < 8; ++r) p[r] = (_Float16)acc[j].e[r];
        } else {         // Q/K [b,h,s,d]
            _Float16* p = (sec == 0 ? qbuf : kbuf) +
                          (((size_t)bb * H_ + head) * S_ + srow) * DH_ + d;
#pragma unroll
            for (int r = 0; r < 8; ++r) p[r * DH_] = (_Float16)acc[j].e[r];
        }
    }
}

// =============================== Kernel 2: flash attention ===============================
// Scores computed TRANSPOSED: S^T[key][q] = K_tile x Q^T, so each lane owns one q
// column and softmax reductions are register ops + one shfl_xor(16).  One softmax
// pass (single alpha) per 32-key block, then P (written to LDS already in [q][key]
// A-layout) times V^T (contiguous global fragments).
constexpr int PSTR = 40;  // P tile 16x32, 80B rows

template <bool MASK>
__device__ __forceinline__ void attn_iter32(const _Float16* __restrict__ K,
                                            const _Float16* __restrict__ Vt, int kb32,
                                            int qb, int rowoff, int col, int lane,
                                            v16h bq0, v16h bq1, _Float16* Pl,
                                            float& mQ, float& lQ, Acc* o) {
    // two 16-key score tiles: D[key][q], key = kb + rowoff + r, q = qb + col
    Acc s0, s1;
    {
        const v16h ak0 = a_frag16(K + (size_t)kb32 * DH_, DH_, lane);
        const v16h ak1 = a_frag16(K + (size_t)kb32 * DH_ + 32, DH_, lane);
        v8f z{};
        s0.v = wmma_f16(ak0, bq0, z);
        s0.v = wmma_f16(ak1, bq1, s0.v);
    }
    {
        const v16h ak0 = a_frag16(K + (size_t)(kb32 + 16) * DH_, DH_, lane);
        const v16h ak1 = a_frag16(K + (size_t)(kb32 + 16) * DH_ + 32, DH_, lane);
        v8f z{};
        s1.v = wmma_f16(ak0, bq0, z);
        s1.v = wmma_f16(ak1, bq1, s1.v);
    }
    // scale + causal mask (key > q); kq = key - q - r
    const int kq = kb32 + rowoff - qb - col;
    float sv0[8], sv1[8];
#pragma unroll
    for (int r = 0; r < 8; ++r) {
        sv0[r] = s0.e[r] * 0.125f;
        sv1[r] = s1.e[r] * 0.125f;
        if (MASK) {
            if (kq + r > 0)      sv0[r] = -1e30f;
            if (kq + 16 + r > 0) sv1[r] = -1e30f;
        }
    }
    // single online-softmax pass over all 32 keys of this q column
    float tmax = sv0[0];
#pragma unroll
    for (int r = 1; r < 8; ++r) tmax = fmaxf(tmax, sv0[r]);
#pragma unroll
    for (int r = 0; r < 8; ++r) tmax = fmaxf(tmax, sv1[r]);
    tmax = fmaxf(tmax, __shfl_xor(tmax, 16, 32));  // combine key halves
    const float mnew  = fmaxf(mQ, tmax);
    const float alpha = __expf(mQ - mnew);
    float p0[8], p1[8], psum = 0.f;
#pragma unroll
    for (int r = 0; r < 8; ++r) { p0[r] = __expf(sv0[r] - mnew); psum += p0[r]; }
#pragma unroll
    for (int r = 0; r < 8; ++r) { p1[r] = __expf(sv1[r] - mnew); psum += p1[r]; }
    psum += __shfl_xor(psum, 16, 32);
    lQ = lQ * alpha + psum;
    mQ = mnew;
    // distribute per-q alpha into the O C-layout (element r -> q-row r+rowoff)
#pragma unroll
    for (int r = 0; r < 8; ++r) {
        const float ar = __shfl(alpha, rowoff + r, 32);
        o[0].e[r] *= ar; o[1].e[r] *= ar; o[2].e[r] *= ar; o[3].e[r] *= ar;
    }
    // store P transposed into [q][key]: two contiguous 8xf16 runs per lane
    _Float16* pp = Pl + col * PSTR + rowoff;
#pragma unroll
    for (int r = 0; r < 8; ++r) pp[r]      = (_Float16)p0[r];
#pragma unroll
    for (int r = 0; r < 8; ++r) pp[16 + r] = (_Float16)p1[r];

    // P @ V: A = P (16 q x 32 keys) from LDS, B = V^T tiles from global
    const v16h ap = a_frag16(Pl, PSTR, lane);
#pragma unroll
    for (int t4 = 0; t4 < 4; ++t4) {
        const v16h bv = b_frag16(Vt + (size_t)(t4 * 16) * S_ + kb32, S_, lane);
        o[t4].v = wmma_f16(ap, bv, o[t4].v);
    }
}

__global__ __launch_bounds__(256) void flash_attn(const _Float16* __restrict__ qbuf,
                                                  const _Float16* __restrict__ kbuf,
                                                  const _Float16* __restrict__ vbuf,
                                                  _Float16* __restrict__ ctx) {
    __shared__ __align__(16) _Float16 smem[16 * PSTR * 8];

    const int t = threadIdx.x, lane = t & 31, wid = t >> 5;
    _Float16* Pl = smem + wid * 16 * PSTR;

    const int bh = blockIdx.y;
    const int qb = blockIdx.x * 128 + wid * 16;
    const size_t base = (size_t)bh * S_ * DH_;
    const _Float16* Q  = qbuf + base;
    const _Float16* K  = kbuf + base;
    const _Float16* Vt = vbuf + base;  // [d][s]

    const int col = lane & 15;
    const int rowoff = (lane >> 4) * 8;

    // Q as B-fragments (col = q, elements along d), loaded once
    const v16h bq0 = b_frag16(Q + (size_t)qb * DH_, DH_, lane);
    const v16h bq1 = b_frag16(Q + (size_t)qb * DH_ + 32, DH_, lane);

    float mQ = -1e30f, lQ = 0.f;  // per-lane (per q column) softmax state
    Acc o[4];
#pragma unroll
    for (int r = 0; r < 8; ++r)
        o[0].e[r] = o[1].e[r] = o[2].e[r] = o[3].e[r] = 0.f;

    const int nk32   = (qb + 16 + 31) >> 5;  // total 32-key iterations
    const int ktFull = qb >> 5;              // fully-unmasked iterations
    int kt = 0;
    for (; kt < ktFull; ++kt)
        attn_iter32<false>(K, Vt, kt * 32, qb, rowoff, col, lane, bq0, bq1, Pl,
                           mQ, lQ, o);
    for (; kt < nk32; ++kt)
        attn_iter32<true>(K, Vt, kt * 32, qb, rowoff, col, lane, bq0, bq1, Pl,
                          mQ, lQ, o);

    // epilogue: per-q 1/l distributed to the O layout, base ptr + immediate offsets
    const int b = bh >> 4, h = bh & 15;
    const float linv = 1.f / lQ;
    _Float16* cp = ctx + ((size_t)(b * S_ + qb + rowoff)) * DOUT + h * DH_ + col;
#pragma unroll
    for (int r = 0; r < 8; ++r) {
        const float inv = __shfl(linv, rowoff + r, 32);
#pragma unroll
        for (int t4 = 0; t4 < 4; ++t4)
            cp[(size_t)r * DOUT + t4 * 16] = (_Float16)(o[t4].e[r] * inv);
    }
}

// =============================== Kernel 3: output projection ===============================
// A-tile (pure f16 copy) staged via the Tensor Data Mover: one D# per wave moves an
// 8-row x 32-col slice into padded LDS rows; completion via s_wait_tensorcnt.
__global__ __launch_bounds__(256) void out_gemm(const _Float16* __restrict__ ctx,
                                                const float* __restrict__ wout,
                                                const float* __restrict__ bout,
                                                float* __restrict__ out) {
    constexpr int LDA = 48;  // 96B rows = 32 f16 data + 16 f16 pad (8 DWORDs)
    constexpr int LDB = 40;
    __shared__ __align__(16) _Float16 sA[64 * LDA];
    __shared__ __align__(16) _Float16 sBt[64 * LDB];  // [n][k]

    const int t = threadIdx.x, lane = t & 31, wid = t >> 5;
    const int nbase = blockIdx.x * 64;
    const int mbase = blockIdx.y * 64;
    const int mt = wid & 3, nw = wid >> 2;

    Acc acc[2];
#pragma unroll
    for (int j = 0; j < 2; ++j)
#pragma unroll
        for (int i = 0; i < 8; ++i) acc[j].e[i] = 0.f;

    for (int k0 = 0; k0 < DOUT; k0 += 32) {
        __syncthreads();
#if __has_builtin(__builtin_amdgcn_tensor_load_to_lds) && \
    __has_builtin(__builtin_amdgcn_s_wait_tensorcnt)
        {   // TDM: wave `wid` DMAs rows [mbase+wid*8, +8) x [k0, k0+32) of ctx into sA
            const unsigned long long ga =
                (unsigned long long)(uintptr_t)(ctx + (size_t)(mbase + wid * 8) * DOUT + k0);
            const unsigned la =
                (unsigned)(uintptr_t)sA + (unsigned)(wid * 8 * LDA * 2);
            v4u_t g0 = { 1u,                                  // count=1, load descriptor
                         la,                                  // lds_addr
                         (unsigned)ga,                        // global_addr[31:0]
                         (unsigned)((ga >> 32) & 0x1ffffffu)  // global_addr[56:32]
                             | (2u << 30) };                  // type=2 (image)
            v8i_t g1 = { (int)((1u << 16)    // data_size = 2B
                             | (1u << 20)    // pad_enable
                             | (3u << 22)    // pad_interval: 16 DWORDs (=64B row)
                             | (7u << 25)),  // pad_amount: 8 DWORDs (=32B pad)
                         (int)((unsigned)DOUT << 16),             // tensor_dim0 lo16
                         (int)(((unsigned)DOUT >> 16) |
                               ((unsigned)(B_ * S_) << 16)),      // dim0 hi | dim1 lo
                         (int)(((unsigned)(B_ * S_) >> 16) |
                               (32u << 16)),                      // dim1 hi | tile_dim0
                         8,                                       // tile_dim1 = 8 rows
                         DOUT,                                    // tensor_dim0_stride
                         0, 0 };
            v4i_t zz = { 0, 0, 0, 0 };
#if defined(__clang_major__) && __clang_major__ >= 23
            v8i_t z8 = { 0, 0, 0, 0, 0, 0, 0, 0 };
            __builtin_amdgcn_tensor_load_to_lds(g0, g1, zz, zz, z8, 0);
#else
            __builtin_amdgcn_tensor_load_to_lds(g0, g1, zz, zz, 0);
#endif
        }
#else
        {   // fallback: manual b128 copy
            const int row = t >> 2, c8 = (t & 3) * 8;
            *(uint4*)(sA + row * LDA + c8) =
                *(const uint4*)(ctx + (size_t)(mbase + row) * DOUT + k0 + c8);
        }
#endif
        // stage B 32k x 64n transposed, coalesced fp32 reads -> f16
#pragma unroll
        for (int it = 0; it < 2; ++it) {
            const int nn = (t & 31) + it * 32;
            const int kg = (t >> 5) * 4;
#pragma unroll
            for (int j = 0; j < 4; ++j)
                sBt[nn * LDB + kg + j] =
                    (_Float16)wout[(size_t)(k0 + kg + j) * DOUT + nbase + nn];
        }
#if __has_builtin(__builtin_amdgcn_tensor_load_to_lds) && \
    __has_builtin(__builtin_amdgcn_s_wait_tensorcnt)
        __builtin_amdgcn_s_wait_tensorcnt(0);
#endif
        __syncthreads();

        const v16h a  = a_frag16(sA + mt * 16 * LDA, LDA, lane);
        const v16h b0 = b_frag16(sBt + (nw * 32 + 0) * LDB, LDB, lane);
        const v16h b1 = b_frag16(sBt + (nw * 32 + 16) * LDB, LDB, lane);
        acc[0].v = wmma_f16(a, b0, acc[0].v);
        acc[1].v = wmma_f16(a, b1, acc[1].v);
    }

    const int col = lane & 15, rowoff = (lane >> 4) * 8;
    const int gm0 = mbase + mt * 16 + rowoff;
#pragma unroll
    for (int j = 0; j < 2; ++j) {
        const int gn = nbase + nw * 32 + j * 16 + col;
        const float bias = bout[gn];
        float* p = out + (size_t)gm0 * DOUT + gn;
#pragma unroll
        for (int r = 0; r < 8; ++r) p[(size_t)r * DOUT] = acc[j].e[r] + bias;
    }
}

// ======================================= launch =======================================
extern "C" void kernel_launch(void* const* d_in, const int* in_sizes, int n_in,
                              void* d_out, int out_size, void* d_ws, size_t ws_size,
                              hipStream_t stream) {
    const float* payload = (const float*)d_in[0];
    const float* wqkv    = (const float*)d_in[1];
    const float* wout    = (const float*)d_in[2];
    const float* bout    = (const float*)d_in[3];
    float*       out     = (float*)d_out;

    const size_t per = (size_t)B_ * H_ * S_ * DH_;
    _Float16* qbuf = (_Float16*)d_ws;
    _Float16* kbuf = qbuf + per;
    _Float16* vbuf = kbuf + per;   // transposed [B,H,DH,S]
    _Float16* ctx  = vbuf + per;   // [B*S, 1024] f16

    dim3 g1(N3 / 64, (B_ * S_) / 64);
    qkv_gemm<<<g1, 256, 0, stream>>>(payload, wqkv, qbuf, kbuf, vbuf);

    dim3 g2(S_ / 128, B_ * H_);
    flash_attn<<<g2, 256, 0, stream>>>(qbuf, kbuf, vbuf, ctx);

    dim3 g3(DOUT / 64, (B_ * S_) / 64);
    out_gemm<<<g3, 256, 0, stream>>>(ctx, wout, bout, out);
}
